// GNN_55705725829229
// MI455X (gfx1250) — matrix operand order repaired and checked by reading, not compiled, not run on hardware
//
#include <hip/hip_runtime.h>
#include <hip/hip_bf16.h>

typedef __attribute__((ext_vector_type(16))) _Float16 v16h;
typedef __attribute__((ext_vector_type(8)))  _Float16 v8h;
typedef __attribute__((ext_vector_type(8)))  float    v8f;
typedef __attribute__((ext_vector_type(4)))  unsigned v4u;
typedef _Float16 h16;

constexpr int kB   = 16;
constexpr int kN   = 114;
constexpr int kF   = 42;    // F1 + F2
constexpr int kF1  = 32;
constexpr int kD   = 100;
constexpr int kDP  = 112;   // D padded to 7 WMMA n-tiles
constexpr int kKP1 = 64;    // F padded (2 k-steps of 32)
constexpr int kKP2 = 128;   // D padded (4 k-steps of 32)
constexpr int kRT  = 207936;              // B*N*N rows
constexpr int kRB  = (kRT + 127) / 128;   // 1625 row blocks
constexpr int kRPad = kRB * 128;          // padded row count (208000)
constexpr int kSG  = 512;   // stats partial grid

union AFrag { unsigned u[8]; v4u q[2]; v16h h; };

// ---------------------------------------------------------------- GEMM 1:
// H0 = relu( XE[:,iter] (R x 42) @ Wi (42 x 100) ), stored f16 padded to 112.
__global__ __launch_bounds__(256) void k_gemm_in(
    const float* __restrict__ XE, const float* __restrict__ Wi,
    h16* __restrict__ H0, int iter)
{
  // As (16384B) + BsF (14336B) = 30720B; Cs (28672B) reuses the same space.
  __shared__ alignas(32) char smem[30720];
  h16*      As  = (h16*)smem;
  unsigned* BsF = (unsigned*)(smem + 128 * kKP1 * 2);  // fragment-major B tile
  h16*      Cs  = (h16*)smem;
  __shared__ int rowOff[128];
  const int t = threadIdx.x;
  const int rowbase = blockIdx.x * 128;

  if (t < 128) {
    int row = rowbase + t, off = -1;
    if (row < kRT) {
      int b = row / (kN * kN);
      int rem = row % (kN * kN);
      int i = rem / kN, j = rem % kN;
      off = (((b * 2 + iter) * kN + i) * kN + j) * kF;
    }
    rowOff[t] = off;
  }
  // B tile fragment-major: BsF[((g*kDP)+n)*8 + r] packs k-pair kp = g*8 + r
  // (g = ks*2 + hi), so one lane's fragment = 8 contiguous DWORDs (2x b128).
  for (int e = t; e < (kKP1 / 2) * kDP; e += 256) {
    int g = e / (8 * kDP);
    int rem = e % (8 * kDP);
    int n = rem >> 3, r = rem & 7;
    int kp = g * 8 + r;          // ks*16 + 8*hi + r
    int k0 = 2 * kp;
    union { h16 h[2]; unsigned u; } pk;
    pk.h[0] = (h16)((k0     < kF && n < kD) ? Wi[k0 * kD + n]       : 0.0f);
    pk.h[1] = (h16)((k0 + 1 < kF && n < kD) ? Wi[(k0 + 1) * kD + n] : 0.0f);
    BsF[e] = pk.u;
  }
  __syncthreads();
  for (int e = t; e < 128 * kKP1; e += 256) {
    int r = e >> 6, k = e & 63;
    int off = rowOff[r];
    As[e] = (h16)((off >= 0 && k < kF) ? XE[off + k] : 0.0f);
  }
  __syncthreads();

  const int lane = t & 31, l16 = lane & 15, hi = lane >> 4;
  const int m0 = (t >> 5) * 16;
  const unsigned* Au = (const unsigned*)As;

  v8f acc[7] = {};
#pragma unroll
  for (int ks = 0; ks < 2; ++ks) {
    AFrag af;
    {
      int base = (m0 + l16) * (kKP1 / 2) + 16 * ks + 8 * hi;
      af.q[0] = *(const v4u*)(Au + base);
      af.q[1] = *(const v4u*)(Au + base + 4);
    }
#pragma unroll
    for (int nt = 0; nt < 7; ++nt) {
      AFrag bf;
      int base = ((ks * 2 + hi) * kDP + nt * 16 + l16) * 8;
      bf.q[0] = *(const v4u*)(BsF + base);
      bf.q[1] = *(const v4u*)(BsF + base + 4);
      acc[nt] = __builtin_amdgcn_wmma_f32_16x16x32_f16(
          false, af.h, false, bf.h, (short)0, acc[nt], false, false);
    }
  }
  __syncthreads();   // As/BsF dead; reuse as Cs
#pragma unroll
  for (int nt = 0; nt < 7; ++nt)
#pragma unroll
    for (int r = 0; r < 8; ++r) {
      float v = acc[nt][r];
      Cs[(m0 + r + 8 * hi) * kDP + nt * 16 + l16] = (h16)(v > 0.f ? v : 0.f);
    }
  __syncthreads();
  // coalesced b128 stores (buffer padded to kRPad rows: no bounds checks)
  const size_t gbase = (size_t)rowbase * kDP;
  for (int c = t; c < 128 * kDP / 8; c += 256)
    *(v8h*)(H0 + gbase + (size_t)c * 8) = *(const v8h*)(Cs + c * 8);
}

// ---------------------------------------------------------------- S[b,q,d] = sum_i f(H[b,i,q,d]) * a[b,i,q]
__global__ __launch_bounds__(128) void k_msum(
    const h16* __restrict__ H, const float* __restrict__ Aadj,
    const float* __restrict__ sc, const float* __restrict__ sh,
    float* __restrict__ S, int iter)
{
  const int bq = blockIdx.x;
  const int b = bq / kN, q = bq % kN;
  const int d = threadIdx.x;
  if (d >= kDP) return;
  const float scl = sc ? sc[d] : 1.f;
  const float shf = sc ? sh[d] : 0.f;
  const bool bn = (sc != nullptr);
  const float* arow = Aadj + (long)(b * 2 + iter) * kN * kN;
  float acc = 0.f;
  for (int i = 0; i < kN; ++i) {
    float av = arow[i * kN + q];
    if (av != 0.f) {
      float hv = (float)H[((long)(b * kN + i) * kN + q) * kDP + d];
      if (bn) { hv = hv * scl + shf; hv = hv > 0.f ? hv : 0.f; }
      acc += hv * av;
    }
  }
  S[(long)(b * kN + q) * kDP + d] = acc;
}

// ---------------------------------------------------------------- GEMM 2/3:
// Xout = H0 + mp(H) @ Wm, mp row (b,i,j) = (S[b,i,:] - f(H[b,j,i,:])) * a[b,i,j]
__global__ __launch_bounds__(256) void k_gemm_mp(
    const float* __restrict__ S, const h16* __restrict__ Hsrc,
    const float* __restrict__ sc, const float* __restrict__ sh,
    const float* __restrict__ Aadj, const float* __restrict__ Wm,
    const h16* __restrict__ H0, h16* __restrict__ Xout, int iter)
{
  // As (32768B) + BsF (28672B) = 61440B; Cs float (57344B) reuses it.
  __shared__ alignas(32) char smem[61440];
  h16*      As  = (h16*)smem;
  unsigned* BsF = (unsigned*)(smem + 128 * kKP2 * 2);
  float*    Cs  = (float*)smem;
  __shared__ float rowA[128];
  __shared__ int   rowS[128];
  __shared__ int   rowH[128];
  const int t = threadIdx.x;
  const int rowbase = blockIdx.x * 128;

  if (t < 128) {
    int row = rowbase + t;
    float aw = 0.f; int so = 0, ho = 0;
    if (row < kRT) {
      int b = row / (kN * kN);
      int rem = row % (kN * kN);
      int i = rem / kN, j = rem % kN;
      aw = Aadj[((b * 2 + iter) * kN + i) * kN + j];
      so = (b * kN + i) * kDP;
      ho = ((b * kN + j) * kN + i) * kDP;
    }
    rowA[t] = aw; rowS[t] = so; rowH[t] = ho;
  }
  for (int e = t; e < (kKP2 / 2) * kDP; e += 256) {
    int g = e / (8 * kDP);
    int rem = e % (8 * kDP);
    int n = rem >> 3, r = rem & 7;
    int kp = g * 8 + r;          // ks*16 + 8*hi + r
    int k0 = 2 * kp;
    union { h16 h[2]; unsigned u; } pk;
    pk.h[0] = (h16)((k0     < kD && n < kD) ? Wm[k0 * kD + n]       : 0.0f);
    pk.h[1] = (h16)((k0 + 1 < kD && n < kD) ? Wm[(k0 + 1) * kD + n] : 0.0f);
    BsF[e] = pk.u;
  }
  __syncthreads();
  const bool bn = (sc != nullptr);
  for (int e = t; e < 128 * kKP2; e += 256) {
    int r = e >> 7, d = e & 127;
    float aw = rowA[r];
    float v = 0.f;
    if (aw != 0.f && d < kD) {     // exploit ~85% adjacency sparsity
      float hv = (float)Hsrc[rowH[r] + d];
      if (bn) { hv = hv * sc[d] + sh[d]; hv = hv > 0.f ? hv : 0.f; }
      v = (S[rowS[r] + d] - hv) * aw;
    }
    As[e] = (h16)v;
  }
  __syncthreads();

  const int lane = t & 31, l16 = lane & 15, hi = lane >> 4;
  const int m0 = (t >> 5) * 16;
  const unsigned* Au = (const unsigned*)As;

  v8f acc[7] = {};
#pragma unroll
  for (int ks = 0; ks < 4; ++ks) {
    AFrag af;
    {
      int base = (m0 + l16) * (kKP2 / 2) + 16 * ks + 8 * hi;
      af.q[0] = *(const v4u*)(Au + base);
      af.q[1] = *(const v4u*)(Au + base + 4);
    }
#pragma unroll
    for (int nt = 0; nt < 7; ++nt) {
      AFrag bf;
      int base = ((ks * 2 + hi) * kDP + nt * 16 + l16) * 8;
      bf.q[0] = *(const v4u*)(BsF + base);
      bf.q[1] = *(const v4u*)(BsF + base + 4);
      acc[nt] = __builtin_amdgcn_wmma_f32_16x16x32_f16(
          false, af.h, false, bf.h, (short)0, acc[nt], false, false);
    }
  }
  __syncthreads();   // As/BsF dead; reuse as float Cs
#pragma unroll
  for (int nt = 0; nt < 7; ++nt)
#pragma unroll
    for (int r = 0; r < 8; ++r)
      Cs[(m0 + r + 8 * hi) * kDP + nt * 16 + l16] = acc[nt][r];
  __syncthreads();
  // coalesced b128 residual-add + store (padded buffers: no bounds checks)
  const size_t gbase = (size_t)rowbase * kDP;
  for (int c = t; c < 128 * kDP / 8; c += 256) {
    int off = c * 8;
    v8h h0v = *(const v8h*)(H0 + gbase + off);
    v8h ov;
#pragma unroll
    for (int i = 0; i < 8; ++i)
      ov[i] = (h16)(Cs[off + i] + (float)h0v[i]);
    *(v8h*)(Xout + gbase + off) = ov;
  }
}

// ---------------------------------------------------------------- BN stats (two-pass, deterministic)
__global__ __launch_bounds__(128) void k_stats_partial(
    const h16* __restrict__ X, float* __restrict__ part)
{
  const int g = blockIdx.x, d = threadIdx.x;
  if (d >= kDP) return;
  const int per = (kRT + kSG - 1) / kSG;
  int r0 = g * per;
  int r1 = r0 + per; if (r1 > kRT) r1 = kRT;
  float s = 0.f, sq = 0.f;
  for (int r = r0; r < r1; ++r) {
    float v = (float)X[(long)r * kDP + d];
    s += v; sq += v * v;
  }
  part[(long)g * (2 * kDP) + d] = s;
  part[(long)g * (2 * kDP) + kDP + d] = sq;
}

__global__ __launch_bounds__(128) void k_stats_final(
    const float* __restrict__ part, const float* __restrict__ gam,
    const float* __restrict__ bet, float* __restrict__ sc, float* __restrict__ sh)
{
  const int d = threadIdx.x;
  if (d >= kDP) return;
  float s = 0.f, sq = 0.f;
  for (int i = 0; i < kSG; ++i) {
    s  += part[(long)i * (2 * kDP) + d];
    sq += part[(long)i * (2 * kDP) + kDP + d];
  }
  if (d < kD) {
    const float inv = 1.0f / (float)kRT;
    float mean = s * inv;
    float var  = sq * inv - mean * mean;
    float scale = gam[d] * rsqrtf(var + 1e-5f);
    sc[d] = scale;
    sh[d] = bet[d] - mean * scale;
  } else { sc[d] = 0.f; sh[d] = 0.f; }   // padded channels stay exactly 0
}

// ---------------------------------------------------------------- readout: Hrow[b,n,:] = relu([x|Mv] @ Wa)
__global__ __launch_bounds__(128) void k_readout(
    const h16* __restrict__ X2, const float* __restrict__ sc, const float* __restrict__ sh,
    const float* __restrict__ Aadj, const float* __restrict__ Xn,
    const float* __restrict__ Wa, float* __restrict__ Hrow, int iter)
{
  __shared__ float XM[kF1 + kD];
  const int bn = blockIdx.x;
  const int b = bn / kN, n = bn % kN;
  const int t = threadIdx.x;
  if (t < kD) {
    const float* arow = Aadj + ((long)(b * 2 + iter) * kN + n) * kN;
    const h16* xrow = X2 + (long)(b * kN + n) * kN * kDP;
    float acc = 0.f;
    for (int k = 0; k < kN; ++k) {
      float av = arow[k];
      if (av != 0.f) {
        float v = (float)xrow[(long)k * kDP + t];
        v = v * sc[t] + sh[t]; v = v > 0.f ? v : 0.f;
        acc += v * av;
      }
    }
    XM[kF1 + t] = acc;
  }
  if (t < kF1)
    XM[t] = Xn[((b * 2 + iter) * kN + n) * kF1 + t];
  __syncthreads();
  if (t < kD) {
    float s = 0.f;
    for (int f = 0; f < kF1 + kD; ++f) s += XM[f] * Wa[f * kD + t];
    Hrow[(long)bn * kD + t] = s > 0.f ? s : 0.f;
  }
}

__global__ __launch_bounds__(128) void k_reduce_max(
    const float* __restrict__ Hrow, float* __restrict__ hout, int first)
{
  const int b = blockIdx.x, d = threadIdx.x;
  if (d >= kD) return;
  float s = 0.f;
  for (int n = 0; n < kN; ++n) s += Hrow[(long)(b * kN + n) * kD + d];
  float* o = hout + b * kD + d;
  *o = first ? s : fmaxf(*o, s);
}

// ---------------------------------------------------------------- head: bn3 + MLP
__global__ __launch_bounds__(256) void k_final(
    const float* __restrict__ hs, const float* __restrict__ hp,
    const float* __restrict__ g3, const float* __restrict__ b3,
    const float* __restrict__ W1, const float* __restrict__ b1v,
    const float* __restrict__ W2, const float* __restrict__ b2v,
    float* __restrict__ out)
{
  __shared__ float H[kB][2 * kD];
  __shared__ float Z[kB][32];
  const int t = threadIdx.x;
  for (int e = t; e < kB * 2 * kD; e += 256) {
    int b = e / (2 * kD), c = e % (2 * kD);
    H[b][c] = (c < kD) ? hs[b * kD + c] : hp[b * kD + (c - kD)];
  }
  __syncthreads();
  if (t < 2 * kD) {
    float s = 0.f, sq = 0.f;
    for (int b = 0; b < kB; ++b) { float v = H[b][t]; s += v; sq += v * v; }
    float mean = s / kB;
    float var  = sq / kB - mean * mean;
    float scale = g3[t] * rsqrtf(var + 1e-5f);
    float shift = b3[t] - mean * scale;
    for (int b = 0; b < kB; ++b) H[b][t] = H[b][t] * scale + shift;
  }
  __syncthreads();
  for (int e = t; e < kB * 32; e += 256) {
    int b = e / 32, u = e % 32;
    float s = b1v[u];
    for (int c = 0; c < 2 * kD; ++c) s += H[b][c] * W1[c * 32 + u];
    Z[b][u] = s > 0.f ? s : 0.f;
  }
  __syncthreads();
  if (t < kB) {
    float s = b2v[0];
    for (int u = 0; u < 32; ++u) s += Z[t][u] * W2[u];
    out[t] = s;
  }
}

extern "C" void kernel_launch(void* const* d_in, const int* in_sizes, int n_in,
                              void* d_out, int out_size, void* d_ws, size_t ws_size,
                              hipStream_t stream) {
  (void)in_sizes; (void)n_in; (void)out_size; (void)ws_size;
  char* ws = (char*)d_ws;
  size_t o = 0;
  auto carve = [&](size_t bytes) -> void* {
    void* p = ws + o;
    o += (bytes + 255) & ~(size_t)255;
    return p;
  };
  h16*   H0   = (h16*)  carve((size_t)kRPad * kDP * sizeof(h16));
  h16*   X1   = (h16*)  carve((size_t)kRPad * kDP * sizeof(h16));
  h16*   X2   = (h16*)  carve((size_t)kRPad * kDP * sizeof(h16));
  float* S    = (float*)carve((size_t)kB * kN * kDP * sizeof(float));
  float* part = (float*)carve((size_t)kSG * 2 * kDP * sizeof(float));
  float* sc1  = (float*)carve(kDP * sizeof(float));
  float* sh1  = (float*)carve(kDP * sizeof(float));
  float* sc2  = (float*)carve(kDP * sizeof(float));
  float* sh2  = (float*)carve(kDP * sizeof(float));
  float* Hrow = (float*)carve((size_t)kB * kN * kD * sizeof(float));
  float* hsub = (float*)carve((size_t)kB * kD * sizeof(float));
  float* hpro = (float*)carve((size_t)kB * kD * sizeof(float));

  for (int branch = 0; branch < 2; ++branch) {
    const float* XE  = (const float*)d_in[branch ? 3 : 0];
    const float* Xn  = (const float*)d_in[branch ? 4 : 1];
    const float* Aa  = (const float*)d_in[branch ? 5 : 2];
    const float* Wi  = (const float*)d_in[branch ? 14 : 6];
    const float* Wm1 = (const float*)d_in[branch ? 15 : 7];
    const float* Wm2 = (const float*)d_in[branch ? 16 : 8];
    const float* Wa  = (const float*)d_in[branch ? 17 : 9];
    const float* g1  = (const float*)d_in[branch ? 18 : 10];
    const float* be1 = (const float*)d_in[branch ? 19 : 11];
    const float* g2  = (const float*)d_in[branch ? 20 : 12];
    const float* be2 = (const float*)d_in[branch ? 21 : 13];
    float* hout = branch ? hpro : hsub;

    for (int iter = 0; iter < 2; ++iter) {
      k_gemm_in<<<kRB, 256, 0, stream>>>(XE, Wi, H0, iter);
      k_msum<<<kB * kN, 128, 0, stream>>>(H0, Aa, nullptr, nullptr, S, iter);
      k_gemm_mp<<<kRB, 256, 0, stream>>>(S, H0, nullptr, nullptr, Aa, Wm1, H0, X1, iter);
      k_stats_partial<<<kSG, 128, 0, stream>>>(X1, part);
      k_stats_final<<<1, 128, 0, stream>>>(part, g1, be1, sc1, sh1);
      k_msum<<<kB * kN, 128, 0, stream>>>(X1, Aa, sc1, sh1, S, iter);
      k_gemm_mp<<<kRB, 256, 0, stream>>>(S, X1, sc1, sh1, Aa, Wm2, H0, X2, iter);
      k_stats_partial<<<kSG, 128, 0, stream>>>(X2, part);
      k_stats_final<<<1, 128, 0, stream>>>(part, g2, be2, sc2, sh2);
      k_readout<<<kB * kN, 128, 0, stream>>>(X2, sc2, sh2, Aa, Xn, Wa, Hrow, iter);
      k_reduce_max<<<kB, 128, 0, stream>>>(Hrow, hout, iter == 0);
    }
  }
  k_final<<<1, 256, 0, stream>>>(hsub, hpro,
      (const float*)d_in[22], (const float*)d_in[23],
      (const float*)d_in[24], (const float*)d_in[25],
      (const float*)d_in[26], (const float*)d_in[27],
      (float*)d_out);
}
